// Attention_87892210745440
// MI455X (gfx1250) — compile-verified, matrix operand order checked by reading, not run on hardware
//
#include <hip/hip_runtime.h>
#include <hip/hip_bf16.h>

// CDNA5 / gfx1250 dual cross-attention pipeline.
// - all matmuls on v_wmma_f32_16x16x32_f16 (wave32)
// - weight / K / V tiles staged into LDS with gfx1250 async global->LDS
//   copies (ASYNCcnt), double-buffered for copy/compute overlap
// - V kept transposed [B,NH,HD,S] so every WMMA fragment read is contiguous

typedef __attribute__((ext_vector_type(16))) _Float16 v16h;
typedef __attribute__((ext_vector_type(8)))  float    v8f;

#define NHEADS 12
#define HD     64
#define CDIM   768
#define SEQ    1024
#define BATCH  8

__device__ __forceinline__ int lane_id() { return threadIdx.x & 31; }

__device__ __forceinline__ v8f wmma_ff(v16h a, v16h b, v8f c) {
  return __builtin_amdgcn_wmma_f32_16x16x32_f16(false, a, false, b, (short)0, c,
                                                false, false);
}

// Async 16-byte global->LDS copy (ASYNCcnt).  vdst = LDS byte address
// (low 32 bits of the shared-aperture flat address), vaddr = global address.
__device__ __forceinline__ void async_b128(unsigned lds_addr, const void* gptr) {
  asm volatile("global_load_async_to_lds_b128 %0, %1, off"
               :: "v"(lds_addr), "v"(gptr) : "memory");
}
__device__ __forceinline__ void wait_async0() {
  asm volatile("s_wait_asynccnt 0x0" ::: "memory");
}

// A/B fragment (16 rows x 32 k) from f16 row-major storage; contiguous along k.
// lanes 0-15: row=lane, K={0..7,16..23}; lanes 16-31: row=lane-16, K={8..15,24..31}.
// Works for global or LDS pointers (compiler picks global_load / ds_load).
__device__ __forceinline__ v16h frag_rm(const _Float16* __restrict__ A, int lda,
                                        int m0, int k0) {
  int l = lane_id();
  const _Float16* p = A + (long)(m0 + (l & 15)) * lda + k0 + ((l >> 4) << 3);
  v16h a;
#pragma unroll
  for (int i = 0; i < 8; ++i) { a[i] = p[i]; a[i + 8] = p[16 + i]; }
  return a;
}

// B fragment with strided columns: element (k,n) at Bm[(k0+k)*ldb + n0+n].
__device__ __forceinline__ v16h frag_cm(const _Float16* __restrict__ Bm, int ldb,
                                        int k0, int n0) {
  int l = lane_id();
  const _Float16* p = Bm + (long)(k0 + ((l >> 4) << 3)) * ldb + n0 + (l & 15);
  v16h b;
#pragma unroll
  for (int i = 0; i < 8; ++i) {
    b[i]     = p[(long)i * ldb];
    b[i + 8] = p[(long)(16 + i) * ldb];
  }
  return b;
}

// ---------------------------------------------------------------- convert
__global__ void cvt_f32_f16(const float* __restrict__ in,
                            _Float16* __restrict__ out, int n) {
  int i = blockIdx.x * blockDim.x + threadIdx.x;
  if (i < n) out[i] = (_Float16)in[i];
}

// ------------------------------------------------------------- ctx projection
__global__ void ctx_proj_kernel(const float* __restrict__ ctx,      // [B,64,1024]
                                const _Float16* __restrict__ Wh,    // [64,768]
                                const float* __restrict__ bias,     // [768]
                                _Float16* __restrict__ out) {       // [8192,768]
  int n0 = blockIdx.x * 16;
  int m0 = blockIdx.y * 16;
  int l = lane_id();
  v8f acc = {};
#pragma unroll
  for (int k0 = 0; k0 < 64; k0 += 32) {
    int m = m0 + (l & 15);
    int b = m >> 10, s = m & 1023;
    int kb = k0 + ((l >> 4) << 3);
    const float* p = ctx + ((long)b * 64 + kb) * 1024 + s;   // transposed read
    v16h a;
#pragma unroll
    for (int i = 0; i < 8; ++i) {
      a[i]     = (_Float16)p[(long)i * 1024];
      a[i + 8] = (_Float16)p[(long)(16 + i) * 1024];
    }
    v16h bf = frag_cm(Wh, CDIM, k0, n0);
    acc = wmma_ff(a, bf, acc);
  }
  int col = n0 + (l & 15);
  int mh = (l < 16) ? 0 : 8;
  float bv = bias[col];
#pragma unroll
  for (int r = 0; r < 8; ++r) {
    int m = m0 + r + mh;
    out[(long)m * CDIM + col] = (_Float16)(acc[r] + bv);
  }
}

// ------------------------------------------------------------- QKV projection
// Y = X[8192,768] @ W[768,768] * scale.  Block tile 32(M)x256(N), 8 waves as
// 2(M)x4(N); each wave computes 16x64 (A fragment reused 4x).  Weight tiles
// double-buffered in LDS via async global->LDS copies (overlap with WMMA).
// transpose_out: 0 -> [B,NH,S,HD] (Q,K);  1 -> [B,NH,HD,S] (V).
__global__ void qkv_proj_kernel(const _Float16* __restrict__ X,
                                const _Float16* __restrict__ W,
                                _Float16* __restrict__ Y,
                                float scale, int transpose_out) {
  __shared__ _Float16 Bs[2 * 32 * 256];                              // 32 KB
  int wave = threadIdx.x >> 5;
  int l = lane_id();
  int n0 = blockIdx.x * 256;
  int m0 = blockIdx.y * 32 + (wave >> 2) * 16;
  int nw = (wave & 3) * 64;
  unsigned lds_base = (unsigned)(size_t)(&Bs[0]);
  v8f acc[4] = {v8f{}, v8f{}, v8f{}, v8f{}};

  auto stage = [&](int buf, int k0) {
#pragma unroll
    for (int i = 0; i < 4; ++i) {
      int idx = threadIdx.x + i * 256;      // 1024 x 16B segments
      int row = idx >> 5;
      int ce  = (idx & 31) << 3;
      async_b128(lds_base + buf * 16384 + idx * 16,
                 W + (long)(k0 + row) * CDIM + n0 + ce);
    }
  };

  stage(0, 0);
  wait_async0();
  __syncthreads();

  for (int kt = 0; kt < 24; ++kt) {
    int k0 = kt * 32;
    int cur = kt & 1;
    if (kt + 1 < 24) stage(cur ^ 1, k0 + 32);     // overlap next copy

    if (kt + 1 < 24)
      __builtin_prefetch(X + (long)(m0 + (l & 15)) * CDIM + k0 + 32, 0, 0);
    v16h a = frag_rm(X, CDIM, m0, k0);
    const _Float16* Bc = &Bs[cur * 32 * 256];
    int kb = (l >> 4) << 3;
    int c0 = nw + (l & 15);
#pragma unroll
    for (int t = 0; t < 4; ++t) {
      v16h bf;
#pragma unroll
      for (int i = 0; i < 8; ++i) {
        bf[i]     = Bc[(kb + i) * 256 + c0 + t * 16];
        bf[i + 8] = Bc[(16 + kb + i) * 256 + c0 + t * 16];
      }
      acc[t] = wmma_ff(a, bf, acc[t]);
    }
    __syncthreads();                               // everyone done with cur
    if (kt + 1 < 24) { wait_async0(); __syncthreads(); }
  }

  int mh = (l < 16) ? 0 : 8;
#pragma unroll
  for (int t = 0; t < 4; ++t) {
    int col = n0 + nw + t * 16 + (l & 15);
    int h = col >> 6, hd = col & 63;
#pragma unroll
    for (int r = 0; r < 8; ++r) {
      int m = m0 + r + mh;
      int b = m >> 10, s = m & 1023;
      _Float16 v = (_Float16)(acc[t][r] * scale);
      if (transpose_out)
        Y[(((long)b * NHEADS + h) * HD + hd) * SEQ + s] = v;   // [B,NH,HD,S]
      else
        Y[(((long)b * NHEADS + h) * SEQ + s) * HD + hd] = v;   // [B,NH,S,HD]
    }
  }
}

// ------------------------------------------------------------------ attention
// 4 waves / block share async-staged, double-buffered K (32x64) and
// V^T (64x32) LDS tiles; each wave owns one 16-query tile.  Flash loop over
// 32-key tiles; Q pre-scaled.  All fragment reads contiguous.
__global__ void attn_kernel(const _Float16* __restrict__ Q,   // [B,NH,S,HD]
                            const _Float16* __restrict__ K,   // [B,NH,S,HD]
                            const _Float16* __restrict__ Vt,  // [B,NH,HD,S]
                            _Float16* __restrict__ O) {       // [B,NH,S,HD]
  __shared__ _Float16 Ks[2][32 * 64];   // 8 KB
  __shared__ _Float16 Vs[2][64 * 32];   // 8 KB
  __shared__ _Float16 P[4][16 * 32];    // 4 KB, per-wave
  int wave = threadIdx.x >> 5;
  int l = lane_id();
  int cl = l & 15;
  int mh = (l < 16) ? 0 : 8;
  int bh = blockIdx.z * NHEADS + blockIdx.y;
  int qt = blockIdx.x * 64 + wave * 16;
  const _Float16* Qb = Q  + (long)bh * SEQ * HD;
  const _Float16* Kb = K  + (long)bh * SEQ * HD;
  const _Float16* Vb = Vt + (long)bh * HD * SEQ;
  _Float16* Ob       = O  + (long)bh * SEQ * HD;
  unsigned ldsK = (unsigned)(size_t)(&Ks[0][0]);
  unsigned ldsV = (unsigned)(size_t)(&Vs[0][0]);
  _Float16* Pw = &P[wave][0];

  auto stage = [&](int buf, int s0) {
    for (int seg = threadIdx.x; seg < 256; seg += 128) {
      int kr = seg >> 3, kc = (seg & 7) << 3;          // K: 32 rows x 128B
      async_b128(ldsK + buf * 4096 + seg * 16, Kb + (long)(s0 + kr) * HD + kc);
      int vr = seg >> 2, vc = (seg & 3) << 3;          // V^T: 64 rows x 64B
      async_b128(ldsV + buf * 4096 + seg * 16, Vb + (long)vr * SEQ + s0 + vc);
    }
  };

  v16h q0 = frag_rm(Qb, HD, qt, 0);
  v16h q1 = frag_rm(Qb, HD, qt, 32);

  float mrow[8], lsum[8];
  v8f o0 = {}, o1 = {}, o2 = {}, o3 = {};
#pragma unroll
  for (int r = 0; r < 8; ++r) { mrow[r] = -1e30f; lsum[r] = 0.f; }

  stage(0, 0);
  wait_async0();
  __syncthreads();

  for (int kt = 0; kt < SEQ / 32; ++kt) {
    int cur = kt & 1;
    if (kt + 1 < SEQ / 32) stage(cur ^ 1, kt * 32 + 32);   // overlap next copy
    const _Float16* Kc = &Ks[cur][0];
    const _Float16* Vc = &Vs[cur][0];

    // scores: Q(16x64) @ K^T(64x32)
    v8f sc0 = {}, sc1 = {};
    sc0 = wmma_ff(q0, frag_rm(Kc, HD, 0, 0),   sc0);
    sc0 = wmma_ff(q1, frag_rm(Kc, HD, 0, 32),  sc0);
    sc1 = wmma_ff(q0, frag_rm(Kc, HD, 16, 0),  sc1);
    sc1 = wmma_ff(q1, frag_rm(Kc, HD, 16, 32), sc1);

    // online softmax (row reductions inside 16-lane halves)
#pragma unroll
    for (int r = 0; r < 8; ++r) {
      float mx = fmaxf(sc0[r], sc1[r]);
#pragma unroll
      for (int off = 8; off >= 1; off >>= 1) mx = fmaxf(mx, __shfl_xor(mx, off, 32));
      float mnew = fmaxf(mrow[r], mx);
      float alpha = __expf(mrow[r] - mnew);
      mrow[r] = mnew;
      float p0 = __expf(sc0[r] - mnew);
      float p1 = __expf(sc1[r] - mnew);
      float rs = p0 + p1;
#pragma unroll
      for (int off = 8; off >= 1; off >>= 1) rs += __shfl_xor(rs, off, 32);
      lsum[r] = lsum[r] * alpha + rs;
      sc0[r] = p0; sc1[r] = p1;
      o0[r] *= alpha; o1[r] *= alpha; o2[r] *= alpha; o3[r] *= alpha;
    }

    // transpose P (C layout -> A-fragment layout) through per-wave LDS
    __syncthreads();
#pragma unroll
    for (int r = 0; r < 8; ++r) {
      Pw[(r + mh) * 32 + cl]      = (_Float16)sc0[r];
      Pw[(r + mh) * 32 + 16 + cl] = (_Float16)sc1[r];
    }
    __syncthreads();
    int kb = (l >> 4) << 3;
    v16h pa;
#pragma unroll
    for (int i = 0; i < 8; ++i) {
      pa[i]     = Pw[cl * 32 + kb + i];
      pa[i + 8] = Pw[cl * 32 + 16 + kb + i];
    }

    // O(16x64) += P(16x32) @ V(32x64); V^T tile rows = d, contiguous in s
    o0 = wmma_ff(pa, frag_rm(Vc, 32, 0, 0),  o0);
    o1 = wmma_ff(pa, frag_rm(Vc, 32, 16, 0), o1);
    o2 = wmma_ff(pa, frag_rm(Vc, 32, 32, 0), o2);
    o3 = wmma_ff(pa, frag_rm(Vc, 32, 48, 0), o3);

    __syncthreads();                                   // done reading cur
    if (kt + 1 < SEQ / 32) { wait_async0(); __syncthreads(); }
  }

#pragma unroll
  for (int r = 0; r < 8; ++r) {
    float inv = 1.f / lsum[r];
    long base = (long)(qt + r + mh) * HD + cl;
    Ob[base]      = (_Float16)(o0[r] * inv);
    Ob[base + 16] = (_Float16)(o1[r] * inv);
    Ob[base + 32] = (_Float16)(o2[r] * inv);
    Ob[base + 48] = (_Float16)(o3[r] * inv);
  }
}

// ---------------------------------------------------------- output projection
// out = O1@Wp + O2@Wp2 + bp + bp2 + x    (f32 out)
__global__ void out_proj_kernel(const _Float16* __restrict__ O1,
                                const _Float16* __restrict__ O2,
                                const _Float16* __restrict__ W1,
                                const _Float16* __restrict__ W2,
                                const float* __restrict__ b1,
                                const float* __restrict__ b2,
                                const float* __restrict__ x,
                                float* __restrict__ out) {
  __shared__ _Float16 B1s[32 * 16];
  __shared__ _Float16 B2s[32 * 16];
  int wave = threadIdx.x >> 5;
  int n0 = blockIdx.x * 16;
  int m0 = (blockIdx.y * 8 + wave) * 16;
  int l = lane_id();
  int row = m0 + (l & 15);
  int bb = row >> 10, s = row & 1023;
  unsigned lds1 = (unsigned)(size_t)(&B1s[0]);
  unsigned lds2 = (unsigned)(size_t)(&B2s[0]);
  v8f acc1 = {}, acc2 = {};
  for (int k0 = 0; k0 < CDIM; k0 += 32) {
    if (threadIdx.x < 64) {
      int idx = threadIdx.x;
      int kr = idx >> 1;
      int ce = (idx & 1) << 3;
      long wi = (long)(k0 + kr) * CDIM + n0 + ce;
      async_b128(lds1 + idx * 16, W1 + wi);
      async_b128(lds2 + idx * 16, W2 + wi);
    }
    wait_async0();
    __syncthreads();

    int h = k0 >> 6;
    int hb = (k0 & 63) + ((l >> 4) << 3);
    const _Float16* p1 = O1 + (((long)bb * NHEADS + h) * SEQ + s) * HD + hb;
    const _Float16* p2 = O2 + (((long)bb * NHEADS + h) * SEQ + s) * HD + hb;
    v16h a1, a2;
#pragma unroll
    for (int i = 0; i < 8; ++i) {
      a1[i] = p1[i]; a1[i + 8] = p1[16 + i];
      a2[i] = p2[i]; a2[i + 8] = p2[16 + i];
    }
    int col = l & 15, kb = (l >> 4) << 3;
    v16h bf1, bf2;
#pragma unroll
    for (int i = 0; i < 8; ++i) {
      bf1[i] = B1s[(kb + i) * 16 + col]; bf1[i + 8] = B1s[(16 + kb + i) * 16 + col];
      bf2[i] = B2s[(kb + i) * 16 + col]; bf2[i + 8] = B2s[(16 + kb + i) * 16 + col];
    }
    acc1 = wmma_ff(a1, bf1, acc1);
    acc2 = wmma_ff(a2, bf2, acc2);
    __syncthreads();
  }
  int col = n0 + (l & 15);
  int mh = (l < 16) ? 0 : 8;
  float bsum = b1[col] + b2[col];
#pragma unroll
  for (int r = 0; r < 8; ++r) {
    long m = m0 + r + mh;
    out[m * CDIM + col] = acc1[r] + acc2[r] + bsum + x[m * CDIM + col];
  }
}

// ------------------------------------------------------------------- launcher
extern "C" void kernel_launch(void* const* d_in, const int* in_sizes, int n_in,
                              void* d_out, int out_size, void* d_ws, size_t ws_size,
                              hipStream_t stream) {
  const float* x    = (const float*)d_in[0];
  const float* ctx  = (const float*)d_in[1];
  const float* Wctx = (const float*)d_in[2];
  const float* bctx = (const float*)d_in[3];
  const float* Wq   = (const float*)d_in[4];
  const float* Wk   = (const float*)d_in[5];
  const float* Wv   = (const float*)d_in[6];
  const float* Wq2  = (const float*)d_in[7];
  const float* Wk2  = (const float*)d_in[8];
  const float* Wv2  = (const float*)d_in[9];
  const float* Wp   = (const float*)d_in[10];
  const float* bp   = (const float*)d_in[11];
  const float* Wp2  = (const float*)d_in[12];
  const float* bp2  = (const float*)d_in[13];
  float* out = (float*)d_out;

  char* ws = (char*)d_ws;
  size_t off = 0;
  auto alloc = [&](size_t bytes) -> void* {
    void* p = (void*)(ws + off);
    off += (bytes + 255) & ~(size_t)255;
    return p;
  };

  const size_t ACT = (size_t)BATCH * SEQ * CDIM;
  const size_t WEL = (size_t)CDIM * CDIM;

  _Float16* xh    = (_Float16*)alloc(ACT * 2);
  _Float16* ctxh  = (_Float16*)alloc(ACT * 2);
  _Float16* Wqh   = (_Float16*)alloc(WEL * 2);
  _Float16* Wkh   = (_Float16*)alloc(WEL * 2);
  _Float16* Wvh   = (_Float16*)alloc(WEL * 2);
  _Float16* Wq2h  = (_Float16*)alloc(WEL * 2);
  _Float16* Wk2h  = (_Float16*)alloc(WEL * 2);
  _Float16* Wv2h  = (_Float16*)alloc(WEL * 2);
  _Float16* Wph   = (_Float16*)alloc(WEL * 2);
  _Float16* Wp2h  = (_Float16*)alloc(WEL * 2);
  _Float16* Wctxh = (_Float16*)alloc((size_t)64 * CDIM * 2);
  _Float16* Q1 = (_Float16*)alloc(ACT * 2);
  _Float16* K1 = (_Float16*)alloc(ACT * 2);
  _Float16* V1 = (_Float16*)alloc(ACT * 2);   // transposed [B,NH,HD,S]
  _Float16* Q2 = (_Float16*)alloc(ACT * 2);
  _Float16* K2 = (_Float16*)alloc(ACT * 2);
  _Float16* V2 = (_Float16*)alloc(ACT * 2);   // transposed [B,NH,HD,S]
  _Float16* O1 = (_Float16*)alloc(ACT * 2);
  _Float16* O2 = (_Float16*)alloc(ACT * 2);

  // downcasts
  cvt_f32_f16<<<(int)((ACT + 255) / 256), 256, 0, stream>>>(x, xh, (int)ACT);
  int nw = (int)WEL, gw = (nw + 255) / 256;
  cvt_f32_f16<<<gw, 256, 0, stream>>>(Wq,  Wqh,  nw);
  cvt_f32_f16<<<gw, 256, 0, stream>>>(Wk,  Wkh,  nw);
  cvt_f32_f16<<<gw, 256, 0, stream>>>(Wv,  Wvh,  nw);
  cvt_f32_f16<<<gw, 256, 0, stream>>>(Wq2, Wq2h, nw);
  cvt_f32_f16<<<gw, 256, 0, stream>>>(Wk2, Wk2h, nw);
  cvt_f32_f16<<<gw, 256, 0, stream>>>(Wv2, Wv2h, nw);
  cvt_f32_f16<<<gw, 256, 0, stream>>>(Wp,  Wph,  nw);
  cvt_f32_f16<<<gw, 256, 0, stream>>>(Wp2, Wp2h, nw);
  cvt_f32_f16<<<(64 * CDIM + 255) / 256, 256, 0, stream>>>(Wctx, Wctxh, 64 * CDIM);

  // context projection (transposed read, K = 64)
  ctx_proj_kernel<<<dim3(CDIM / 16, (BATCH * SEQ) / 16), 32, 0, stream>>>(
      ctx, Wctxh, bctx, ctxh);

  // QKV projections (Q pre-scaled by 1/sqrt(HD) = 0.125; V stored transposed)
  const float SC = 0.125f;
  dim3 ggrid(CDIM / 256, (BATCH * SEQ) / 32);   // (3, 256)
  qkv_proj_kernel<<<ggrid, 256, 0, stream>>>(ctxh, Wqh,  Q1, SC,  0);
  qkv_proj_kernel<<<ggrid, 256, 0, stream>>>(xh,   Wkh,  K1, 1.f, 0);
  qkv_proj_kernel<<<ggrid, 256, 0, stream>>>(xh,   Wvh,  V1, 1.f, 1);
  qkv_proj_kernel<<<ggrid, 256, 0, stream>>>(xh,   Wq2h, Q2, SC,  0);
  qkv_proj_kernel<<<ggrid, 256, 0, stream>>>(ctxh, Wk2h, K2, 1.f, 0);
  qkv_proj_kernel<<<ggrid, 256, 0, stream>>>(ctxh, Wv2h, V2, 1.f, 1);

  // flash attention, both branches: 4 waves/block share staged K/V tiles
  dim3 agrid(SEQ / 64, NHEADS, BATCH);          // (16, 12, 8)
  attn_kernel<<<agrid, 128, 0, stream>>>(Q1, K1, V1, O1);
  attn_kernel<<<agrid, 128, 0, stream>>>(Q2, K2, V2, O2);

  // fused output projections + biases + residual
  dim3 pgrid(CDIM / 16, (BATCH * SEQ) / (16 * 8));
  out_proj_kernel<<<pgrid, 256, 0, stream>>>(O1, O2, Wph, Wp2h, bp, bp2, x, out);
}
